// CrossModalAttention_1494648619442
// MI455X (gfx1250) — matrix operand order repaired and checked by reading, not compiled, not run on hardware
//
#include <hip/hip_runtime.h>
#include <hip/hip_bf16.h>
#include <cstddef>
#include <cstdint>

// ---------------------------------------------------------------------------
// CrossModalAttention for MI455X (gfx1250, wave32, WMMA 16x16x32 f16,
// async global->LDS staging per CDNA5 ISA)
// B=4, C=256, H=W=48 -> N=2304 tokens, 8 heads, head_dim=32, SCALE=1/sqrt(32)
// ---------------------------------------------------------------------------

#define BATCH   4
#define CDIM    256
#define NTOK    2304          // 48*48
#define NHEAD   8
#define HDIM    32
#define SCALE   0.17677669529663689f   // 32^-0.5
#define LSTRIDE 40            // LDS row stride in halves (80B: 16B-aligned rows,
                              // row-to-row bank step 20 -> conflict-free in 16 lanes)

typedef __attribute__((ext_vector_type(16))) _Float16 v16h;
typedef __attribute__((ext_vector_type(8)))  _Float16 h8;
typedef __attribute__((ext_vector_type(8)))  float    v8f;

union FragU { v16h v; h8 h[2]; };

__device__ __forceinline__ v8f wmma_f16(v16h a, v16h b, v8f c) {
    // D = A(16x32 f16) * B(32x16 f16) + C(16x16 f32)
    return __builtin_amdgcn_wmma_f32_16x16x32_f16(
        false, a, false, b, (short)0, c, false, false);
}

// -------- async global -> LDS copy (ASYNCcnt-tracked DMA, ISA ch.10/§15.18) --
__device__ __forceinline__ void async_g2l_b128(uint32_t lds_off, const void* gaddr) {
    // VDST = LDS byte offset VGPR, VADDR = 64-bit global address VGPR pair
    asm volatile("global_load_async_to_lds_b128 %0, %1, off"
                 :: "v"(lds_off), "v"(gaddr) : "memory");
}
__device__ __forceinline__ void wait_async0() {
    asm volatile("s_wait_asynccnt 0x0" ::: "memory");
}
__device__ __forceinline__ uint32_t lds_off32(const void* p) {
    // flat LDS aperture keeps the group-segment byte offset in addr[31:0]
    return (uint32_t)(uintptr_t)p;
}

// A-fragment (16x32 f16), row-major source, row stride in halves.
// ISA §7.12.2: lanes 0-15 -> M=lane, K in {0..7,16..23}; lanes 16-31 ->
// M=lane-16, K in {8..15,24..31}.  Two 16B loads per lane.
__device__ __forceinline__ v16h load_frag_a(const _Float16* base, int stride) {
    const int lane = threadIdx.x & 31;
    const int m    = lane & 15;
    const int kb   = (lane < 16) ? 0 : 8;
    const _Float16* row = base + (size_t)m * stride;
    FragU u;
    u.h[0] = *(const h8*)(row + kb);
    u.h[1] = *(const h8*)(row + kb + 16);
    return u.v;
}

// B-fragment (32x16 f16) from [n][k] source (row = output col, contiguous K).
// lanes 0-15 -> N=lane, K in {0..7,16..23}; lanes 16-31 -> N=lane-16, K in {8..15,24..31}.
__device__ __forceinline__ v16h load_frag_bt(const _Float16* base, int stride) {
    const int lane = threadIdx.x & 31;
    const int n    = lane & 15;
    const int kb   = (lane < 16) ? 0 : 8;
    const _Float16* row = base + (size_t)n * stride;
    FragU u;
    u.h[0] = *(const h8*)(row + kb);
    u.h[1] = *(const h8*)(row + kb + 16);
    return u.v;
}

__device__ __forceinline__ float redmax16(float x) {
    x = fmaxf(x, __shfl_xor(x, 8, 32));
    x = fmaxf(x, __shfl_xor(x, 4, 32));
    x = fmaxf(x, __shfl_xor(x, 2, 32));
    x = fmaxf(x, __shfl_xor(x, 1, 32));
    return x;
}
__device__ __forceinline__ float redsum16(float x) {
    x += __shfl_xor(x, 8, 32);
    x += __shfl_xor(x, 4, 32);
    x += __shfl_xor(x, 2, 32);
    x += __shfl_xor(x, 1, 32);
    return x;
}

// ---------------------------------------------------------------------------
// Kernel 1: QKV projection.  X is (B, C, N) channel-major fp32; logical GEMM
// (B*N x C) @ (C x Cout).  fp32 source needs conversion, so tiles are staged
// through registers (all loads in flight) then cvt+ds_store.  Q mode
// (outV==null): Q[bh][n][d].  KV mode: cols<256 -> K[bh][n][d], cols>=256 ->
// V^T[bh][d][n].  Block 128 threads / 4 waves; tile 64x64, K-step 32.
// ---------------------------------------------------------------------------
__global__ void cm_proj_qkv(const float* __restrict__ X,
                            const float* __restrict__ W, int Cout,
                            _Float16* __restrict__ outQK,
                            _Float16* __restrict__ outV) {
    __shared__ alignas(16) _Float16 lds_a[64 * LSTRIDE];
    __shared__ alignas(16) _Float16 lds_b[64 * LSTRIDE];

    const int tid  = threadIdx.x;
    const int lane = tid & 31;
    const int wave = tid >> 5;

    const int row_base = blockIdx.x * 64;
    const int col_base = blockIdx.y * 64;
    const int b  = row_base / NTOK;                  // 64 | 2304 -> no straddle
    const int n0 = row_base - b * NTOK;

    v8f acc[4] = {{}, {}, {}, {}};

    for (int k0 = 0; k0 < CDIM; k0 += 32) {
        __syncthreads();
        // A tile: 64 tokens x 32 channels.  512 float4 vectors, 4 per thread,
        // all loads issued before conversion (one loadcnt wait).
        float4 av[4], bv[4];
        #pragma unroll
        for (int i = 0; i < 4; ++i) {
            int vi = tid + i * 128;
            int nn = (vi & 15) * 4, kk = vi >> 4;
            av[i] = *(const float4*)&X[(size_t)b * CDIM * NTOK
                                       + (size_t)(k0 + kk) * NTOK + n0 + nn];
            bv[i] = *(const float4*)&W[(size_t)(k0 + kk) * Cout + col_base + nn];
        }
        #pragma unroll
        for (int i = 0; i < 4; ++i) {
            int vi = tid + i * 128;
            int nn = (vi & 15) * 4, kk = vi >> 4;
            lds_a[(nn + 0) * LSTRIDE + kk] = (_Float16)av[i].x;
            lds_a[(nn + 1) * LSTRIDE + kk] = (_Float16)av[i].y;
            lds_a[(nn + 2) * LSTRIDE + kk] = (_Float16)av[i].z;
            lds_a[(nn + 3) * LSTRIDE + kk] = (_Float16)av[i].w;
            lds_b[(nn + 0) * LSTRIDE + kk] = (_Float16)bv[i].x;
            lds_b[(nn + 1) * LSTRIDE + kk] = (_Float16)bv[i].y;
            lds_b[(nn + 2) * LSTRIDE + kk] = (_Float16)bv[i].z;
            lds_b[(nn + 3) * LSTRIDE + kk] = (_Float16)bv[i].w;
        }
        __syncthreads();

        v16h bf = load_frag_bt(&lds_b[wave * 16 * LSTRIDE], LSTRIDE);
        #pragma unroll
        for (int mm = 0; mm < 4; ++mm) {
            v16h af = load_frag_a(&lds_a[mm * 16 * LSTRIDE], LSTRIDE);
            acc[mm] = wmma_f16(af, bf, acc[mm]);
        }
    }

    // C-fragment: VGPR r -> (M=r, N=lane) lanes<16, (M=r+8, N=lane-16) lanes>=16
    const int col = col_base + wave * 16 + (lane & 15);
    const int hi  = (lane >= 16) ? 8 : 0;
    // wave-uniform mode select (col block is 16-aligned, so col>=256 is uniform)
    if (outV != nullptr && (col_base + wave * 16) >= 256) {
        const int c2 = col - 256;
        const int h = c2 >> 5, d = c2 & 31;
        _Float16* dst = outV + ((size_t)(b * NHEAD + h) * HDIM + d) * NTOK + n0;
        #pragma unroll
        for (int mm = 0; mm < 4; ++mm)
            #pragma unroll
            for (int r = 0; r < 8; ++r)
                dst[mm * 16 + r + hi] = (_Float16)acc[mm][r];
    } else {
        const int h = col >> 5, d = col & 31;
        _Float16* dst = outQK + ((size_t)(b * NHEAD + h) * NTOK + n0) * HDIM + d;
        #pragma unroll
        for (int mm = 0; mm < 4; ++mm)
            #pragma unroll
            for (int r = 0; r < 8; ++r)
                dst[(size_t)(mm * 16 + r + hi) * HDIM] = (_Float16)acc[mm][r];
    }
}

// ---------------------------------------------------------------------------
// Kernel 2: flash attention per (b,h).  4 waves/block share (b,h) and stage
// the common 32-key K/V tiles into LDS with double-buffered async DMA copies;
// each wave owns one 16-query tile.  Per 32-key step: 2 score WMMAs + online
// softmax + P relayout through wave-private LDS + 2 PV WMMAs.
// ---------------------------------------------------------------------------
__global__ void cm_attention(const _Float16* __restrict__ Q,
                             const _Float16* __restrict__ K,
                             const _Float16* __restrict__ Vt,
                             const float* __restrict__ um, int use_uw,
                             _Float16* __restrict__ att) {
    __shared__ alignas(16) _Float16 lds_k[2][32 * LSTRIDE];   // [key][d]
    __shared__ alignas(16) _Float16 lds_v[2][32 * LSTRIDE];   // [d][key]
    __shared__ alignas(16) _Float16 lds_p[4][16 * LSTRIDE];   // P relayout

    const int tid  = threadIdx.x;
    const int lane = tid & 31;
    const int wave = tid >> 5;
    const int bh   = blockIdx.y;
    const int b    = bh >> 3;
    const int h    = bh & 7;
    const int q0   = (blockIdx.x * 4 + wave) * 16;
    const int hi   = (lane >= 16) ? 8 : 0;

    _Float16* P = lds_p[wave];

    const _Float16* Qb = Q  + ((size_t)bh * NTOK + q0) * HDIM;
    const _Float16* Kb = K  + (size_t)bh * NTOK * HDIM;
    const _Float16* Vb = Vt + (size_t)bh * HDIM * NTOK;

    const v16h qf = load_frag_a(Qb, HDIM);

    float uwv[8];
    #pragma unroll
    for (int r = 0; r < 8; ++r) {
        uwv[r] = use_uw ? (SCALE / (um[(size_t)b * NTOK + q0 + r + hi] + 1e-6f))
                        : SCALE;
    }

    float m_i[8], l_i[8];
    #pragma unroll
    for (int r = 0; r < 8; ++r) { m_i[r] = -3.0e38f; l_i[r] = 0.0f; }
    v8f O0 = {}, O1 = {};

    // async stage of one 32-key tile: 128 threads x (16B K + 16B V)
    const int srow = tid >> 2;        // 0..31
    const int sch  = (tid & 3) * 8;   // half offset of 16B chunk
    auto stage = [&](int kb, int buf) {
        async_g2l_b128(lds_off32(&lds_k[buf][srow * LSTRIDE + sch]),
                       Kb + (size_t)(kb + srow) * HDIM + sch);
        async_g2l_b128(lds_off32(&lds_v[buf][srow * LSTRIDE + sch]),
                       Vb + (size_t)srow * NTOK + kb + sch);
    };

    stage(0, 0);

    for (int kb = 0; kb < NTOK; kb += 32) {
        const int cur = (kb >> 5) & 1;
        wait_async0();          // this wave's copies for tile `cur` complete
        __syncthreads();        // all waves' copies visible; prev compute done
        if (kb + 32 < NTOK) stage(kb + 32, cur ^ 1);   // overlap next DMA

        // scores: S = Q (16x32) x K^T (32x16) from staged [key][d] tile
        v8f s0acc = {}, s1acc = {};
        {
            v16h k0f = load_frag_bt(&lds_k[cur][0],             LSTRIDE);
            v16h k1f = load_frag_bt(&lds_k[cur][16 * LSTRIDE],  LSTRIDE);
            s0acc = wmma_f16(qf, k0f, s0acc);
            s1acc = wmma_f16(qf, k1f, s1acc);
        }

        // online softmax (row r+hi lives in one 16-lane half-wave)
        const int col = lane & 15;
        #pragma unroll
        for (int r = 0; r < 8; ++r) {
            float s0 = s0acc[r] * uwv[r];
            float s1 = s1acc[r] * uwv[r];
            float tm = redmax16(fmaxf(s0, s1));
            float mn = fmaxf(m_i[r], tm);
            float p0 = __expf(s0 - mn);
            float p1 = __expf(s1 - mn);
            float rs = redsum16(p0 + p1);
            float al = __expf(m_i[r] - mn);
            l_i[r] = l_i[r] * al + rs;
            m_i[r] = mn;
            O0[r] *= al;
            O1[r] *= al;
            P[(r + hi) * LSTRIDE + col]      = (_Float16)p0;
            P[(r + hi) * LSTRIDE + col + 16] = (_Float16)p1;
        }

        // O += P (16x32) x V (32x32) from staged [d][key] tile
        v16h pf = load_frag_a(P, LSTRIDE);
        {
            v16h v0f = load_frag_bt(&lds_v[cur][0],            LSTRIDE);
            v16h v1f = load_frag_bt(&lds_v[cur][16 * LSTRIDE], LSTRIDE);
            O0 = wmma_f16(pf, v0f, O0);
            O1 = wmma_f16(pf, v1f, O1);
        }
    }

    const int dcol = lane & 15;
    #pragma unroll
    for (int r = 0; r < 8; ++r) {
        float inv = 1.0f / l_i[r];
        size_t base = ((size_t)b * NTOK + q0 + r + hi) * CDIM + h * HDIM;
        att[base + dcol]      = (_Float16)(O0[r] * inv);
        att[base + dcol + 16] = (_Float16)(O1[r] * inv);
    }
}

// ---------------------------------------------------------------------------
// Kernel 3: Y = att (B*N x 256) @ Wp (256x256) + bp -> (B, C, N) fp32.
// att is f16 so the shared A tile is staged with async DMA; W is fp32 and
// staged through registers with conversion.
// ---------------------------------------------------------------------------
__global__ void cm_proj_out(const _Float16* __restrict__ A,
                            const float* __restrict__ W,
                            const float* __restrict__ bias,
                            float* __restrict__ out) {
    __shared__ alignas(16) _Float16 lds_a[64 * LSTRIDE];
    __shared__ alignas(16) _Float16 lds_b[64 * LSTRIDE];

    const int tid  = threadIdx.x;
    const int lane = tid & 31;
    const int wave = tid >> 5;

    const int row_base = blockIdx.x * 64;
    const int col_base = blockIdx.y * 64;
    const int b  = row_base / NTOK;
    const int n0 = row_base - b * NTOK;

    v8f acc[4] = {{}, {}, {}, {}};

    // A tile is 64 rows x 32 halves (64B/row) = 256 x 16B chunks, 2 per thread
    const int ar0 = tid >> 1;                 // rows 0..63
    const int ac0 = (tid & 1) * 16;           // half-offset 0 / 16

    for (int k0 = 0; k0 < CDIM; k0 += 32) {
        __syncthreads();      // previous compute done before overwrite
        // async-stage f16 activation tile
        async_g2l_b128(lds_off32(&lds_a[ar0 * LSTRIDE + ac0]),
                       A + (size_t)(row_base + ar0) * CDIM + k0 + ac0);
        async_g2l_b128(lds_off32(&lds_a[ar0 * LSTRIDE + ac0 + 8]),
                       A + (size_t)(row_base + ar0) * CDIM + k0 + ac0 + 8);
        // cvt-stage fp32 weight tile
        float4 bv[4];
        #pragma unroll
        for (int i = 0; i < 4; ++i) {
            int vi = tid + i * 128;
            int cc = (vi & 15) * 4, kk = vi >> 4;
            bv[i] = *(const float4*)&W[(size_t)(k0 + kk) * CDIM + col_base + cc];
        }
        #pragma unroll
        for (int i = 0; i < 4; ++i) {
            int vi = tid + i * 128;
            int cc = (vi & 15) * 4, kk = vi >> 4;
            lds_b[(cc + 0) * LSTRIDE + kk] = (_Float16)bv[i].x;
            lds_b[(cc + 1) * LSTRIDE + kk] = (_Float16)bv[i].y;
            lds_b[(cc + 2) * LSTRIDE + kk] = (_Float16)bv[i].z;
            lds_b[(cc + 3) * LSTRIDE + kk] = (_Float16)bv[i].w;
        }
        wait_async0();
        __syncthreads();

        v16h bf = load_frag_bt(&lds_b[wave * 16 * LSTRIDE], LSTRIDE);
        #pragma unroll
        for (int mm = 0; mm < 4; ++mm) {
            v16h af = load_frag_a(&lds_a[mm * 16 * LSTRIDE], LSTRIDE);
            acc[mm] = wmma_f16(af, bf, acc[mm]);
        }
    }

    const int col = col_base + wave * 16 + (lane & 15);
    const int hi  = (lane >= 16) ? 8 : 0;
    const float bvs = bias[col];
    float* dst = out + ((size_t)b * CDIM + col) * NTOK + n0;
    #pragma unroll
    for (int mm = 0; mm < 4; ++mm)
        #pragma unroll
        for (int r = 0; r < 8; ++r)
            dst[mm * 16 + r + hi] = acc[mm][r] + bvs;
}

// ---------------------------------------------------------------------------
// Launch
// ---------------------------------------------------------------------------
extern "C" void kernel_launch(void* const* d_in, const int* in_sizes, int n_in,
                              void* d_out, int out_size, void* d_ws, size_t ws_size,
                              hipStream_t stream) {
    (void)in_sizes; (void)n_in; (void)out_size; (void)ws_size;

    const float* img  = (const float*)d_in[0];
    const float* rad  = (const float*)d_in[1];
    const float* um   = (const float*)d_in[2];
    const float* Wq_img    = (const float*)d_in[3];
    const float* Wkv_radar = (const float*)d_in[4];
    const float* Wq_radar  = (const float*)d_in[5];
    const float* Wkv_img   = (const float*)d_in[6];
    const float* Wp_img    = (const float*)d_in[7];
    const float* bp_img    = (const float*)d_in[8];
    const float* Wp_radar  = (const float*)d_in[9];
    const float* bp_radar  = (const float*)d_in[10];
    float* out = (float*)d_out;

    _Float16* ws = (_Float16*)d_ws;
    const size_t S = (size_t)BATCH * NHEAD * NTOK * HDIM;   // 2,359,296 halves
    _Float16* Qi = ws + 0 * S;
    _Float16* Kr = ws + 1 * S;
    _Float16* Vr = ws + 2 * S;   // transposed [bh][d][n]
    _Float16* Qr = ws + 3 * S;
    _Float16* Ki = ws + 4 * S;
    _Float16* Vi = ws + 5 * S;   // transposed
    _Float16* Ai = ws + 6 * S;   // att img  (B,N,C) f16
    _Float16* Ar = ws + 7 * S;   // att radar

    dim3 blk(128);
    const int MT = (BATCH * NTOK) / 64;   // 144 M-tiles

    cm_proj_qkv<<<dim3(MT, 4), blk, 0, stream>>>(img, Wq_img,    256, Qi, nullptr);
    cm_proj_qkv<<<dim3(MT, 8), blk, 0, stream>>>(rad, Wkv_radar, 512, Kr, Vr);
    cm_proj_qkv<<<dim3(MT, 4), blk, 0, stream>>>(rad, Wq_radar,  256, Qr, nullptr);
    cm_proj_qkv<<<dim3(MT, 8), blk, 0, stream>>>(img, Wkv_img,   512, Ki, Vi);

    cm_attention<<<dim3(36, 32), blk, 0, stream>>>(Qi, Kr, Vr, um, 1, Ai);
    cm_attention<<<dim3(36, 32), blk, 0, stream>>>(Qr, Ki, Vi, nullptr, 0, Ar);

    cm_proj_out<<<dim3(MT, 4), blk, 0, stream>>>(Ai, Wp_img,   bp_img,   out);
    cm_proj_out<<<dim3(MT, 4), blk, 0, stream>>>(Ar, Wp_radar, bp_radar,
                                                 out + (size_t)BATCH * CDIM * NTOK);
}
